// QLinear_12128987644267
// MI455X (gfx1250) — matrix operand order repaired and checked by reading, not compiled, not run on hardware
//
#include <hip/hip_runtime.h>
#include <hip/hip_bf16.h>

// ---------------------------------------------------------------------------
// QLinear: out[m,n] = SCALE * sum_k x[m,k]*w[n,k] + bias[n]
// M=8192 (B*S), N=8192, K=2048, fp32 in/out.
//
// MI455X / gfx1250 strategy:
//  - bf16 hi/lo split of both operands, 3-term product (xh*wh + xh*wl + xl*wh)
//    accumulated fp32 via v_wmma_f32_16x16x32_bf16 (~fp32 accuracy, bf16 speed).
//  - 128x128 block tile, BK=32, DOUBLE-BUFFERED LDS filled with
//    GLOBAL_LOAD_ASYNC_TO_LDS_B128 (ASYNCcnt) -> copy of tile k+1 overlaps the
//    24 WMMAs of tile k, one barrier per K-step, no VGPR staging.
//  - Non-temporal output stores: keep the 128MB of bf16 operands L2-resident
//    (192MB L2), stream the 256MB output past it.
// ---------------------------------------------------------------------------

#ifndef __has_builtin
#define __has_builtin(x) 0
#endif

#if __has_builtin(__builtin_amdgcn_global_load_async_to_lds_b128)
#define HAVE_ASYNC_LDS 1
#else
#define HAVE_ASYNC_LDS 0
#endif

typedef __attribute__((ext_vector_type(16))) __bf16 bf16x16;
typedef __attribute__((ext_vector_type(8)))  float  floatx8;
typedef __attribute__((ext_vector_type(4)))  float  f32x4;
typedef __attribute__((ext_vector_type(4)))  int    i32x4;

#if HAVE_ASYNC_LDS
typedef __attribute__((address_space(1))) i32x4 gi32x4;   // global int4
typedef __attribute__((address_space(3))) i32x4 li32x4;   // LDS int4
#endif

union FragU {
    bf16x16 v;
    uint4   u[2];   // two 16-byte halves
};

// 16B global -> LDS copy, per-lane. Async (ASYNCcnt) when available.
__device__ __forceinline__ void cp16_async(const unsigned short* g, unsigned short* l) {
#if HAVE_ASYNC_LDS
    __builtin_amdgcn_global_load_async_to_lds_b128(
        (gi32x4*)g, (li32x4*)l, /*offset=*/0, /*cpol=*/0);
#else
    *(uint4*)l = *(const uint4*)g;
#endif
}

__device__ __forceinline__ void wait_async0() {
#if HAVE_ASYNC_LDS
#if __has_builtin(__builtin_amdgcn_s_wait_asynccnt)
    __builtin_amdgcn_s_wait_asynccnt(0);
#else
    asm volatile("s_wait_asynccnt 0x0" ::: "memory");
#endif
#endif
}

// ---------------- fp32 -> (bf16 hi, bf16 lo) split (RNE) --------------------
__device__ __forceinline__ void split1(float x, unsigned short& h, unsigned short& l) {
    unsigned int u = __float_as_uint(x);
    unsigned int r = u + 0x7FFFu + ((u >> 16) & 1u);       // round-to-nearest-even
    h = (unsigned short)(r >> 16);
    float hf = __uint_as_float((unsigned int)h << 16);
    float d  = x - hf;                                      // exact residual
    unsigned int u2 = __float_as_uint(d);
    unsigned int r2 = u2 + 0x7FFFu + ((u2 >> 16) & 1u);
    l = (unsigned short)(r2 >> 16);
}

__global__ void split_bf16_kernel(const float* __restrict__ src,
                                  unsigned short* __restrict__ hi,
                                  unsigned short* __restrict__ lo,
                                  size_t n4) {
    size_t i = (size_t)blockIdx.x * blockDim.x + threadIdx.x;
    if (i >= n4) return;
    f32x4 f = __builtin_nontemporal_load((const f32x4*)src + i);  // read-once stream
    ushort4 h, l;
    split1(f[0], h.x, l.x);
    split1(f[1], h.y, l.y);
    split1(f[2], h.z, l.z);
    split1(f[3], h.w, l.w);
    ((ushort4*)hi)[i] = h;   // re-read many times -> default (L2-resident)
    ((ushort4*)lo)[i] = l;
}

// ------------------------------- WMMA GEMM ---------------------------------
#define BM 128
#define BN 128
#define BK 32
#define LDS_STRIDE 40   // 32 + 8-elem pad: 80B rows, 16B aligned, bank-friendly
#define TILE_ELEMS (BM * LDS_STRIDE)

__global__ __launch_bounds__(256)
void qlinear_wmma_kernel(const unsigned short* __restrict__ Ah,
                         const unsigned short* __restrict__ Al,
                         const unsigned short* __restrict__ Bh,
                         const unsigned short* __restrict__ Bl,
                         const float* __restrict__ bias,
                         float* __restrict__ out,
                         int M, int N, int K) {
    __shared__ unsigned short sAh[2][TILE_ELEMS];
    __shared__ unsigned short sAl[2][TILE_ELEMS];
    __shared__ unsigned short sBh[2][TILE_ELEMS];
    __shared__ unsigned short sBl[2][TILE_ELEMS];

    const int tid  = threadIdx.x;
    const int lane = tid & 31;
    const int wid  = tid >> 5;       // 0..7
    const int wm   = wid >> 1;       // 0..3 : 32 rows each
    const int wn   = wid & 1;        // 0..1 : 64 cols each

    const int bm = blockIdx.y * BM;
    const int bn = blockIdx.x * BN;

    // A fragment (16-bit 16x32): lanes 0-15 -> K {0..7,16..23}; 16-31 -> {8..15,24..31}
    const int kbA = (lane < 16) ? 0 : 8;
    // B fragment (32x16): lanes 0-15 -> K {0..7,16..23}; 16-31 -> {8..15,24..31}
    const int kbB = (lane < 16) ? 0 : 8;
    const int halfLane = lane & 15;

    floatx8 acc[2][4];
#pragma unroll
    for (int mi = 0; mi < 2; ++mi)
#pragma unroll
        for (int ni = 0; ni < 4; ++ni)
            acc[mi][ni] = (floatx8)0.0f;

    const int nk = K / BK;

    // ---- issue async copies for the first K-tile into buffer 0 ----
#pragma unroll
    for (int p = 0; p < 2; ++p) {
        int c   = tid + p * 256;     // 512 chunks of 16B per tile
        int row = c >> 2;            // 0..127
        int off = (c & 3) * 8;       // 0,8,16,24 elements
        size_t gA = (size_t)(bm + row) * K + off;
        size_t gB = (size_t)(bn + row) * K + off;
        int loff  = row * LDS_STRIDE + off;
        cp16_async(Ah + gA, &sAh[0][loff]);
        cp16_async(Al + gA, &sAl[0][loff]);
        cp16_async(Bh + gB, &sBh[0][loff]);
        cp16_async(Bl + gB, &sBl[0][loff]);
    }

    for (int kt = 0; kt < nk; ++kt) {
        const int cur = kt & 1;

        // our async writes into buf[cur] done, then block-wide visibility
        wait_async0();
        __syncthreads();

        // kick off next tile into the other buffer (overlaps WMMAs below)
        if (kt + 1 < nk) {
            const int k1 = (kt + 1) * BK;
            const int nxt = cur ^ 1;
#pragma unroll
            for (int p = 0; p < 2; ++p) {
                int c   = tid + p * 256;
                int row = c >> 2;
                int off = (c & 3) * 8;
                size_t gA = (size_t)(bm + row) * K + k1 + off;
                size_t gB = (size_t)(bn + row) * K + k1 + off;
                int loff  = row * LDS_STRIDE + off;
                cp16_async(Ah + gA, &sAh[nxt][loff]);
                cp16_async(Al + gA, &sAl[nxt][loff]);
                cp16_async(Bh + gB, &sBh[nxt][loff]);
                cp16_async(Bl + gB, &sBl[nxt][loff]);
            }
        }

        // ---- fragments from LDS ----
        FragU ah[2], al[2], bh[4], bl[4];
#pragma unroll
        for (int mi = 0; mi < 2; ++mi) {
            int r = wm * 32 + mi * 16 + halfLane;
            ah[mi].u[0] = *(const uint4*)&sAh[cur][r * LDS_STRIDE + kbA];
            ah[mi].u[1] = *(const uint4*)&sAh[cur][r * LDS_STRIDE + kbA + 16];
            al[mi].u[0] = *(const uint4*)&sAl[cur][r * LDS_STRIDE + kbA];
            al[mi].u[1] = *(const uint4*)&sAl[cur][r * LDS_STRIDE + kbA + 16];
        }
#pragma unroll
        for (int ni = 0; ni < 4; ++ni) {
            int r = wn * 64 + ni * 16 + halfLane;
            bh[ni].u[0] = *(const uint4*)&sBh[cur][r * LDS_STRIDE + kbB];
            bh[ni].u[1] = *(const uint4*)&sBh[cur][r * LDS_STRIDE + kbB + 16];
            bl[ni].u[0] = *(const uint4*)&sBl[cur][r * LDS_STRIDE + kbB];
            bl[ni].u[1] = *(const uint4*)&sBl[cur][r * LDS_STRIDE + kbB + 16];
        }

        // ---- 3-term bf16 product, fp32 accumulate: 24 WMMAs per K-step ----
#pragma unroll
        for (int mi = 0; mi < 2; ++mi) {
#pragma unroll
            for (int ni = 0; ni < 4; ++ni) {
                acc[mi][ni] = __builtin_amdgcn_wmma_f32_16x16x32_bf16(
                    false, ah[mi].v, false, bh[ni].v, (short)0, acc[mi][ni], false, false);
                acc[mi][ni] = __builtin_amdgcn_wmma_f32_16x16x32_bf16(
                    false, ah[mi].v, false, bl[ni].v, (short)0, acc[mi][ni], false, false);
                acc[mi][ni] = __builtin_amdgcn_wmma_f32_16x16x32_bf16(
                    false, al[mi].v, false, bh[ni].v, (short)0, acc[mi][ni], false, false);
            }
        }
    }

    // ---- epilogue: D layout (lanes 0-15: M=r, N=lane; 16-31: M=8+r, N=lane-16)
    const float scale = 0.0078125f;
    const int rowOff  = (lane >> 4) * 8;
    const int col     = halfLane;
    const int gm_base = bm + wm * 32 + rowOff;
    const int gn_base = bn + wn * 64 + col;
#pragma unroll
    for (int ni = 0; ni < 4; ++ni) {
        float bv = bias[gn_base + ni * 16];
#pragma unroll
        for (int mi = 0; mi < 2; ++mi) {
            float* p = out + (size_t)(gm_base + mi * 16) * N + (gn_base + ni * 16);
#pragma unroll
            for (int r = 0; r < 8; ++r) {
                __builtin_nontemporal_store(scale * acc[mi][ni][r] + bv, p);
                p += N;
            }
        }
    }
}

// ------------------------------- launcher ----------------------------------
extern "C" void kernel_launch(void* const* d_in, const int* in_sizes, int n_in,
                              void* d_out, int out_size, void* d_ws, size_t ws_size,
                              hipStream_t stream) {
    const float* x    = (const float*)d_in[0];   // [M, K] flattened (B*S, K)
    const float* w    = (const float*)d_in[1];   // [N, K]
    const float* bias = (const float*)d_in[2];   // [N]
    float* out = (float*)d_out;

    const int N = in_sizes[2];
    const int K = in_sizes[1] / N;
    const int M = in_sizes[0] / K;

    const size_t MK = (size_t)M * K;
    const size_t NK = (size_t)N * K;

    unsigned short* xh = (unsigned short*)d_ws;
    unsigned short* xl = xh + MK;
    unsigned short* wh = xl + MK;
    unsigned short* wl = wh + NK;

    {
        size_t n4 = MK / 4;
        split_bf16_kernel<<<dim3((unsigned)((n4 + 255) / 256)), dim3(256), 0, stream>>>(
            x, xh, xl, n4);
    }
    {
        size_t n4 = NK / 4;
        split_bf16_kernel<<<dim3((unsigned)((n4 + 255) / 256)), dim3(256), 0, stream>>>(
            w, wh, wl, n4);
    }

    dim3 grid(N / BN, M / BM);
    qlinear_wmma_kernel<<<grid, dim3(256), 0, stream>>>(xh, xl, wh, wl, bias, out, M, N, K);
}
